// Attention_42674795053691
// MI455X (gfx1250) — matrix operand order repaired and checked by reading, not compiled, not run on hardware
//
#include <hip/hip_runtime.h>
#include <hip/hip_bf16.h>
#include <math.h>

#define BATCH 64
#define SEQ   2048
#define HID   1024
#define ASTR  1032   // padded LDS row stride (bf16 elems): 2064 B, 16B-aligned, bank-spread

typedef __attribute__((ext_vector_type(16))) __bf16 v16bf;
typedef __attribute__((ext_vector_type(8)))  float  v8f;

union FragB16 {
    v16bf bf;
    unsigned int u[8];
};

__device__ __forceinline__ unsigned int f32_to_bf16_rne(float f) {
    unsigned int u = __float_as_uint(f);
    u += 0x7FFFu + ((u >> 16) & 1u);   // round-to-nearest-even
    return u >> 16;
}

// ---------------------------------------------------------------------------
// Kernel 0: W_h (f32, [N=1024][K=1024] row-major as W_h[o][h]) -> bf16 in
// WMMA B-fragment layout. Fragment (n_tile, ks) = 16(N) x 32(K) block, stored
// as 256 dwords: dword index = lane*8 + d, lane<16 -> K base 0, lane>=16 -> K
// base 16; dword d packs K = base+2d (lo) and base+2d+1 (hi), N = n_tile*16 +
// (lane&15).  Hot-loop loads become two coalesced 16B loads per lane.
// ---------------------------------------------------------------------------
__global__ __launch_bounds__(256) void swizzle_wh_kernel(
    const float* __restrict__ Wh, unsigned int* __restrict__ wswz)
{
    int t    = blockIdx.x * 256 + threadIdx.x;   // dword index, 0 .. 512K-1
    int frag = t >> 8;                           // / 256 dwords per fragment
    int rem  = t & 255;
    int l    = rem >> 3;                         // lane 0..31
    int d    = rem & 7;                          // dword within lane
    int ntile = frag >> 5;                       // 0..63
    int ks    = frag & 31;                       // 0..31
    int n = ntile * 16 + (l & 15);
    int k = ks * 32 + ((l >> 4) << 4) + 2 * d;
    const float* p = Wh + (size_t)n * HID + k;
    unsigned int lo = f32_to_bf16_rne(p[0]);
    unsigned int hv = f32_to_bf16_rne(p[1]);
    wswz[t] = lo | (hv << 16);
}

// ---------------------------------------------------------------------------
// Kernel 1: dec_proj[b][o] = sum_h dec[b][h] * W_s[o][h]   (tiny: 134 MFLOP)
// ---------------------------------------------------------------------------
__global__ __launch_bounds__(256) void dec_proj_kernel(
    const float* __restrict__ dec, const float* __restrict__ Ws,
    float* __restrict__ dproj)
{
    __shared__ float hvec[HID];
    int b   = blockIdx.x >> 2;
    int oc  = blockIdx.x & 3;
    int tid = threadIdx.x;
    const float* db = dec + (size_t)b * HID;
    for (int i = tid; i < HID; i += 256) hvec[i] = db[i];
    __syncthreads();
    int o = oc * 256 + tid;
    const float* wr = Ws + (size_t)o * HID;
    float s = 0.f;
    #pragma unroll 4
    for (int k = 0; k < HID; k += 4) {
        float4 w4 = *(const float4*)(wr + k);
        s += w4.x * hvec[k] + w4.y * hvec[k + 1] + w4.z * hvec[k + 2] + w4.w * hvec[k + 3];
    }
    dproj[(size_t)b * HID + o] = s;
}

// ---------------------------------------------------------------------------
// Kernel 2 (hot): per block = one (b, 16-row s-tile).
//   enc tile -> bf16 in LDS (once), 8 waves each own a 128-wide N strip:
//   8 accum tiles x 32 k-steps of v_wmma_f32_16x16x32_bf16 (256 WMMA / wave),
//   then fused  score[s] = sum_o v[o]*tanh(enc_proj + dec_proj).
// ---------------------------------------------------------------------------
__global__ __launch_bounds__(256) void attn_scores_wmma_kernel(
    const float* __restrict__ enc, const unsigned int* __restrict__ wswz,
    const float* __restrict__ dproj, const float* __restrict__ v,
    float* __restrict__ scores)
{
    __shared__ __align__(16) unsigned short Abuf[16 * ASTR];
    __shared__ float scpart[8][16];

    const int tid = threadIdx.x;
    const int b   = blockIdx.x >> 7;       // / (SEQ/16)
    const int st  = blockIdx.x & 127;
    const int s0  = st << 4;

    // Stage A tile (16 x 1024) as bf16, coalesced f32 reads.
    const float* arow = enc + ((size_t)b * SEQ + s0) * HID;
    #pragma unroll 4
    for (int i = tid; i < 16 * HID; i += 256) {
        int m = i >> 10, k = i & (HID - 1);
        Abuf[m * ASTR + k] = (unsigned short)f32_to_bf16_rne(arow[(size_t)m * HID + k]);
    }
    __syncthreads();

    const int lane = tid & 31;
    const int wv   = tid >> 5;
    const int lm   = lane & 15;
    const int hi   = lane >> 4;
    const int nbase = wv << 7;                      // wave covers N in [nbase, nbase+128)

    v8f acc[8] = {};
    const unsigned int* bbase = wswz + (size_t)lane * 8;

    for (int ks = 0; ks < 32; ++ks) {
        // A fragment (contiguous K to match B): lane<16 -> K ks*32+{0..15},
        // lane>=16 -> +16.  Two ds_load_b128 per lane.
        const unsigned short* ap = &Abuf[lm * ASTR + ks * 32 + hi * 16];
        uint4 a0 = *(const uint4*)(ap);
        uint4 a1 = *(const uint4*)(ap + 8);
        FragB16 a;
        a.u[0] = a0.x; a.u[1] = a0.y; a.u[2] = a0.z; a.u[3] = a0.w;
        a.u[4] = a1.x; a.u[5] = a1.y; a.u[6] = a1.z; a.u[7] = a1.w;

        #pragma unroll
        for (int nt = 0; nt < 8; ++nt) {
            int frag = ((nbase >> 4) + nt) * 32 + ks;
            const unsigned int* bp = bbase + (size_t)frag * 256;
            uint4 b0 = *(const uint4*)(bp);
            uint4 b1 = *(const uint4*)(bp + 4);
            FragB16 bb;
            bb.u[0] = b0.x; bb.u[1] = b0.y; bb.u[2] = b0.z; bb.u[3] = b0.w;
            bb.u[4] = b1.x; bb.u[5] = b1.y; bb.u[6] = b1.z; bb.u[7] = b1.w;
            acc[nt] = __builtin_amdgcn_wmma_f32_16x16x32_bf16(
                false, a.bf, false, bb.bf, (short)0, acc[nt], false, false);
        }
    }

    // Epilogue.  C layout: VGPR r, lane<16 -> (M=r, N=lane); lane>=16 -> (M=r+8).
    float part[8] = {0.f, 0.f, 0.f, 0.f, 0.f, 0.f, 0.f, 0.f};
    #pragma unroll
    for (int nt = 0; nt < 8; ++nt) {
        int n = nbase + nt * 16 + lm;
        float dp = dproj[(size_t)b * HID + n];
        float vn = v[n];
        #pragma unroll
        for (int r = 0; r < 8; ++r) {
            float e = tanhf(acc[nt][r] + dp);
            part[r] += vn * e;
        }
    }
    // Reduce over the 16 lanes of each half (xor masks < 16 stay in-half).
    #pragma unroll
    for (int r = 0; r < 8; ++r)
        for (int m = 1; m < 16; m <<= 1)
            part[r] += __shfl_xor(part[r], m, 32);

    if (lane == 0) {
        for (int r = 0; r < 8; ++r) scpart[wv][r] = part[r];
    }
    if (lane == 16) {
        for (int r = 0; r < 8; ++r) scpart[wv][8 + r] = part[r];
    }
    __syncthreads();

    if (tid < 16) {
        float s = 0.f;
        #pragma unroll
        for (int w2 = 0; w2 < 8; ++w2) s += scpart[w2][tid];
        scores[(size_t)b * SEQ + s0 + tid] = s;
    }
}

// ---------------------------------------------------------------------------
// Kernel 3: row softmax over S=2048 per batch.
// ---------------------------------------------------------------------------
__global__ __launch_bounds__(256) void softmax_kernel(
    const float* __restrict__ scores, float* __restrict__ weights)
{
    __shared__ float red[256];
    int b = blockIdx.x, tid = threadIdx.x;
    const float* sr = scores + (size_t)b * SEQ;
    float loc[8];
    float mx = -INFINITY;
    #pragma unroll
    for (int j = 0; j < 8; ++j) { loc[j] = sr[tid + j * 256]; mx = fmaxf(mx, loc[j]); }
    red[tid] = mx; __syncthreads();
    for (int off = 128; off; off >>= 1) {
        if (tid < off) red[tid] = fmaxf(red[tid], red[tid + off]);
        __syncthreads();
    }
    mx = red[0]; __syncthreads();
    float sum = 0.f;
    #pragma unroll
    for (int j = 0; j < 8; ++j) { loc[j] = __expf(loc[j] - mx); sum += loc[j]; }
    red[tid] = sum; __syncthreads();
    for (int off = 128; off; off >>= 1) {
        if (tid < off) red[tid] += red[tid + off];
        __syncthreads();
    }
    float inv = 1.0f / red[0];
    float* wr = weights + (size_t)b * SEQ;
    #pragma unroll
    for (int j = 0; j < 8; ++j) wr[tid + j * 256] = loc[j] * inv;
}

// ---------------------------------------------------------------------------
// Kernel 4: context[b][h] = sum_s weights[b][s] * enc[b][s][h]  (memory-bound,
// coalesced; weights staged in LDS).  grid = (64, 4), block handles 256 h's.
// ---------------------------------------------------------------------------
__global__ __launch_bounds__(256) void context_kernel(
    const float* __restrict__ enc, const float* __restrict__ weights,
    float* __restrict__ ctx)
{
    __shared__ float wbuf[SEQ];
    int b = blockIdx.x, tid = threadIdx.x;
    int h = blockIdx.y * 256 + tid;
    const float* wr = weights + (size_t)b * SEQ;
    for (int i = tid; i < SEQ; i += 256) wbuf[i] = wr[i];
    __syncthreads();
    const float* eb = enc + (size_t)b * SEQ * HID + h;
    float acc = 0.f;
    #pragma unroll 4
    for (int s = 0; s < SEQ; ++s)
        acc += wbuf[s] * eb[(size_t)s * HID];
    ctx[(size_t)b * HID + h] = acc;
}

// ---------------------------------------------------------------------------
extern "C" void kernel_launch(void* const* d_in, const int* in_sizes, int n_in,
                              void* d_out, int out_size, void* d_ws, size_t ws_size,
                              hipStream_t stream)
{
    const float* dec = (const float*)d_in[0];   // [64,1,1024]
    const float* enc = (const float*)d_in[1];   // [64,2048,1024]
    const float* W_h = (const float*)d_in[2];   // [1024,1024]
    const float* W_s = (const float*)d_in[3];   // [1024,1024]
    const float* v   = (const float*)d_in[4];   // [1024]

    float* ctx_out = (float*)d_out;                    // context: 64*1024
    float* w_out   = (float*)d_out + BATCH * HID;      // weights: 64*2048

    unsigned int* wswz = (unsigned int*)d_ws;                            // 2 MB
    float* dproj  = (float*)((char*)d_ws + (2u << 20));                  // 256 KB
    float* scores = (float*)((char*)d_ws + (2u << 20) + BATCH * HID * 4);// 512 KB

    swizzle_wh_kernel<<<(HID * HID / 2) / 256, 256, 0, stream>>>(W_h, wswz);
    dec_proj_kernel<<<BATCH * 4, 256, 0, stream>>>(dec, W_s, dproj);
    attn_scores_wmma_kernel<<<BATCH * (SEQ / 16), 256, 0, stream>>>(enc, wswz, dproj, v, scores);
    softmax_kernel<<<BATCH, 256, 0, stream>>>(scores, w_out);
    context_kernel<<<dim3(BATCH, 4), 256, 0, stream>>>(enc, w_out, ctx_out);
}
